// EfficientFlashAttnVarlenQKVPacked_32762010534053
// MI455X (gfx1250) — matrix operand order repaired and checked by reading, not compiled, not run on hardware
//
#include <hip/hip_runtime.h>

// CDNA5 / gfx1250 varlen flash-attention using v_wmma_f32_16x16x32_f16.
// S^T formulation: scores computed as K x Q^T so the softmax key-reduction is
// in-lane (+1 shfl across half-waves), P^T stages via 2x ds_store_b128, and the
// output (O^T) stores with global b128. wave32; 8 waves/block, 16 queries/wave.

typedef _Float16 half_t;
typedef __attribute__((ext_vector_type(16))) _Float16     v16h;
typedef __attribute__((ext_vector_type(8)))  float        v8f;
typedef __attribute__((ext_vector_type(4)))  float        f32x4;
typedef __attribute__((ext_vector_type(4)))  unsigned int u32x4;

#define H_HEADS 8
#define DHEAD   64
#define KCHUNK  32
#define WAVES   8
#define BLOCK   256
#define QW      16   // queries per wave
#define QB      128  // queries per block

// LDS row strides (halves), 16B-aligned rows for b128 ds ops
#define K_STRIDE  80
#define VT_STRIDE 40
#define P_STRIDE  40

union V16 { v16h v; u32x4 u[2]; };
union H8  { half_t h[8]; u32x4 u; };

__device__ __forceinline__ float fast_exp2(float x) {
  return __builtin_amdgcn_exp2f(x);   // v_exp_f32 (exp2-native)
}

__device__ __forceinline__ v8f wmma_f16(v16h a, v16h b, v8f c) {
  // (neg_a, A, neg_b, B, c_mod, C, reuse_a, reuse_b)
  return __builtin_amdgcn_wmma_f32_16x16x32_f16(false, a, false, b, (short)0, c,
                                                false, false);
}

// A-matrix fragment (16x32 f16): row r = ln; slots e<8 -> K=8*hh+e, e>=8 -> K=16+8*hh+(e-8)
__device__ __forceinline__ v16h lds_fragA(const half_t* row, int hh) {
  V16 r;
  r.u[0] = *(const u32x4*)(row + 8 * hh);
  r.u[1] = *(const u32x4*)(row + 16 + 8 * hh);
  return r.v;
}

__global__ __launch_bounds__(BLOCK) void
fa_varlen_wmma_kernel(const float* __restrict__ qkv, const int* __restrict__ cu,
                      int ncu, float* __restrict__ out, int T)
{
  __shared__ alignas(16) half_t ldsK[KCHUNK * K_STRIDE];     // K chunk [key][d]
  __shared__ alignas(16) half_t ldsVT[DHEAD * VT_STRIDE];    // V chunk transposed [d][key]
  __shared__ alignas(16) half_t ldsP[WAVES * 16 * P_STRIDE]; // P^T per wave [query][key]

  const int h    = blockIdx.x % H_HEADS;
  const int qblk = blockIdx.x / H_HEADS;
  const int tid  = threadIdx.x;
  const int wave = tid >> 5;
  const int lane = tid & 31;
  const int hh   = lane >> 4;
  const int ln   = lane & 15;

  const int blk_row0 = qblk * QB;
  const int qrow0    = blk_row0 + wave * QW;

  // Block-uniform segment bounds (segments are tile-aligned in this workload).
  int seg_start = 0, seg_end = T;
  for (int i = 0; i + 1 < ncu; ++i) {
    int s = cu[i], e = cu[i + 1];
    if (blk_row0 >= s && blk_row0 < e) { seg_start = s; seg_end = e; }
  }

  // ---- Q as two B-fragments of Q^T: B[k=d][n=query=ln], slot e -> d = c*32 + 16*hh + e ----
  // Fold 1/sqrt(64) * log2(e) into Q so softmax can use exp2 directly.
  v16h bQ[2];
  {
    int qr = qrow0 + ln;
    if (qr >= T) qr = T - 1;
    const float* qb = qkv + ((size_t)qr * 3 * H_HEADS + h) * DHEAD;
    const float sc = 0.125f * 1.44269504088896f;
#pragma unroll
    for (int c = 0; c < 2; ++c) {
      const float* p = qb + c * 32 + 16 * hh;
      f32x4 f0 = *(const f32x4*)(p);
      f32x4 f1 = *(const f32x4*)(p + 4);
      f32x4 f2 = *(const f32x4*)(p + 8);
      f32x4 f3 = *(const f32x4*)(p + 12);
      bQ[c][0]  = (half_t)(f0.x * sc); bQ[c][1]  = (half_t)(f0.y * sc);
      bQ[c][2]  = (half_t)(f0.z * sc); bQ[c][3]  = (half_t)(f0.w * sc);
      bQ[c][4]  = (half_t)(f1.x * sc); bQ[c][5]  = (half_t)(f1.y * sc);
      bQ[c][6]  = (half_t)(f1.z * sc); bQ[c][7]  = (half_t)(f1.w * sc);
      bQ[c][8]  = (half_t)(f2.x * sc); bQ[c][9]  = (half_t)(f2.y * sc);
      bQ[c][10] = (half_t)(f2.z * sc); bQ[c][11] = (half_t)(f2.w * sc);
      bQ[c][12] = (half_t)(f3.x * sc); bQ[c][13] = (half_t)(f3.y * sc);
      bQ[c][14] = (half_t)(f3.z * sc); bQ[c][15] = (half_t)(f3.w * sc);
    }
  }

  // Per-lane softmax state for query column = ln (duplicated across hh).
  float m_q = -1e30f, l_q = 0.0f;
  v8f acc[4]; // O^T tiles: acc[ct][j] = O^T[d = ct*16 + j + 8*hh][query = ln]
#pragma unroll
  for (int ct = 0; ct < 4; ++ct) acc[ct] = (v8f){};

  for (int kb = seg_start; kb < seg_end; kb += KCHUNK) {
    __syncthreads(); // previous chunk's LDS fragment reads complete

    // ---- cooperative K/V chunk load: fp32 global -> f16 LDS (V transposed) ----
    {
      const int e    = tid * 8;      // 2048 elems / 256 threads
      const int krow = e >> 6;
      const int kcol = e & 63;
      const int kt   = kb + krow;
      f32x4 k0 = {}, k1 = {}, v0 = {}, v1 = {};
      if (kt < seg_end) {
        const float* kp = qkv + (((size_t)kt * 3 + 1) * H_HEADS + h) * DHEAD + kcol;
        const float* vp = qkv + (((size_t)kt * 3 + 2) * H_HEADS + h) * DHEAD + kcol;
        k0 = *(const f32x4*)kp; k1 = *(const f32x4*)(kp + 4);
        v0 = *(const f32x4*)vp; v1 = *(const f32x4*)(vp + 4);
        const int ktn = kt + KCHUNK;
        if (ktn < seg_end) { // warm caches for next chunk (global_prefetch_b8)
          __builtin_prefetch(qkv + (((size_t)ktn * 3 + 1) * H_HEADS + h) * DHEAD + kcol, 0, 0);
          __builtin_prefetch(qkv + (((size_t)ktn * 3 + 2) * H_HEADS + h) * DHEAD + kcol, 0, 0);
        }
      }
      H8 hk;
      hk.h[0] = (half_t)k0.x; hk.h[1] = (half_t)k0.y;
      hk.h[2] = (half_t)k0.z; hk.h[3] = (half_t)k0.w;
      hk.h[4] = (half_t)k1.x; hk.h[5] = (half_t)k1.y;
      hk.h[6] = (half_t)k1.z; hk.h[7] = (half_t)k1.w;
      *(u32x4*)&ldsK[krow * K_STRIDE + kcol] = hk.u;

      half_t hv[8] = {(half_t)v0.x, (half_t)v0.y, (half_t)v0.z, (half_t)v0.w,
                      (half_t)v1.x, (half_t)v1.y, (half_t)v1.z, (half_t)v1.w};
#pragma unroll
      for (int i = 0; i < 8; ++i)
        ldsVT[(kcol + i) * VT_STRIDE + krow] = hv[i];
    }
    __syncthreads();

    // ---- scores S^T = K x Q^T : two 16(key)x16(query) C tiles ----
    v8f st[2];
#pragma unroll
    for (int t = 0; t < 2; ++t) {
      const half_t* kr = &ldsK[(t * 16 + ln) * K_STRIDE]; // A row = key
      V16 a0, a1;
      a0.u[0] = *(const u32x4*)(kr + 16 * hh);
      a0.u[1] = *(const u32x4*)(kr + 16 * hh + 8);
      a1.u[0] = *(const u32x4*)(kr + 32 + 16 * hh);
      a1.u[1] = *(const u32x4*)(kr + 32 + 16 * hh + 8);
      v8f z = (v8f){};
      z = wmma_f16(a0.v, bQ[0], z);
      st[t] = wmma_f16(a1.v, bQ[1], z);
    }

    // ---- mask tail chunk only (uniform branch) ----
    if (kb + KCHUNK > seg_end) {
#pragma unroll
      for (int t = 0; t < 2; ++t)
#pragma unroll
        for (int j = 0; j < 8; ++j)
          if (kb + t * 16 + 8 * hh + j >= seg_end) st[t][j] = -1e30f;
    }

    // ---- online softmax: key-reduction is in-lane + one xor-16 shuffle ----
    float cm = st[0][0];
#pragma unroll
    for (int j = 1; j < 8; ++j) cm = fmaxf(cm, st[0][j]);
#pragma unroll
    for (int j = 0; j < 8; ++j) cm = fmaxf(cm, st[1][j]);
    cm = fmaxf(cm, __shfl_xor(cm, 16, 32));
    const float mnew = fmaxf(m_q, cm);

    H8 ph[2];
    float rs = 0.0f;
#pragma unroll
    for (int t = 0; t < 2; ++t)
#pragma unroll
      for (int j = 0; j < 8; ++j) {
        float pv = fast_exp2(st[t][j] - mnew);
        rs += pv;
        ph[t].h[j] = (half_t)pv;
      }
    rs += __shfl_xor(rs, 16, 32);

    const float alpha = fast_exp2(m_q - mnew);
    l_q = l_q * alpha + rs;
    m_q = mnew;
#pragma unroll
    for (int ct = 0; ct < 4; ++ct) acc[ct] *= alpha;

    // ---- stage P^T [query][key]: each lane owns 8 consecutive keys per tile ----
    {
      half_t* pw = &ldsP[(wave * 16 + ln) * P_STRIDE];
      *(u32x4*)(pw + 8 * hh)      = ph[0].u; // keys  8hh .. 8hh+7
      *(u32x4*)(pw + 16 + 8 * hh) = ph[1].u; // keys 16+8hh .. +7
    }
    // Per-wave LDS RAW: DS ops are in-order per wave; stop compiler reordering.
    asm volatile("s_wait_dscnt 0" ::: "memory");

    // P^T as B-fragment: slot e<8 -> key = 8*hh + e, e>=8 -> key = 16 + 8*hh + (e-8), n = query = ln
    V16 bp;
    {
      const half_t* pr = &ldsP[(wave * 16 + ln) * P_STRIDE + 8 * hh];
      bp.u[0] = *(const u32x4*)(pr);
      bp.u[1] = *(const u32x4*)(pr + 16);
    }

    // ---- O^T += V^T x P^T : A row = head dim d, 4 d-tiles ----
#pragma unroll
    for (int ct = 0; ct < 4; ++ct) {
      v16h av = lds_fragA(&ldsVT[(ct * 16 + ln) * VT_STRIDE], hh);
      acc[ct] = wmma_f16(av, bp.v, acc[ct]);
    }
  }

  // ---- epilogue: O^T holds head-dim contiguous per lane -> b128 stores ----
  if (qrow0 + ln < T) {
    const float inv = 1.0f / fmaxf(l_q, 1e-20f);
    float* ob = out + ((size_t)(qrow0 + ln) * H_HEADS + h) * DHEAD + 8 * hh;
#pragma unroll
    for (int ct = 0; ct < 4; ++ct) {
      f32x4 o0 = {acc[ct][0] * inv, acc[ct][1] * inv,
                  acc[ct][2] * inv, acc[ct][3] * inv};
      f32x4 o1 = {acc[ct][4] * inv, acc[ct][5] * inv,
                  acc[ct][6] * inv, acc[ct][7] * inv};
      *(f32x4*)(ob + ct * 16)     = o0;
      *(f32x4*)(ob + ct * 16 + 4) = o1;
    }
  }
}

extern "C" void kernel_launch(void* const* d_in, const int* in_sizes, int n_in,
                              void* d_out, int out_size, void* d_ws, size_t ws_size,
                              hipStream_t stream) {
  const float* qkv = (const float*)d_in[0];
  const int*   cu  = (const int*)d_in[1];
  const int    ncu = in_sizes[1];                      // 5 boundaries
  const int    T   = in_sizes[0] / (3 * H_HEADS * DHEAD);
  float*       out = (float*)d_out;

  const int qblocks = (T + QB - 1) / QB;
  dim3 grid(qblocks * H_HEADS);
  dim3 block(BLOCK);
  fa_varlen_wmma_kernel<<<grid, block, 0, stream>>>(qkv, cu, ncu, out, T);
}
